// MLP_27650999451937
// MI455X (gfx1250) — 8-bit matrix operand order repaired and checked by reading, compiled, not run on hardware
//
#include <hip/hip_runtime.h>
#include <hip/hip_bf16.h>
#include <stdint.h>

typedef int v8i_t __attribute__((ext_vector_type(8)));

#define QMAXF 127.0f
#define EPSQ  1e-8f

__device__ __forceinline__ float gelu_exact(float x) {
    return 0.5f * x * (1.0f + erff(x * 0.70710678118654752440f));
}

__global__ void absmax_kernel(const float* __restrict__ x, size_t n4,
                              unsigned* __restrict__ out_bits) {
    float m = 0.0f;
    const float4* x4 = (const float4*)x;
    size_t stride = (size_t)gridDim.x * blockDim.x;
    for (size_t i = (size_t)blockIdx.x * blockDim.x + threadIdx.x; i < n4; i += stride) {
        float4 v = x4[i];
        m = fmaxf(m, fmaxf(fmaxf(fabsf(v.x), fabsf(v.y)), fmaxf(fabsf(v.z), fabsf(v.w))));
    }
    for (int off = 16; off > 0; off >>= 1)
        m = fmaxf(m, __shfl_xor(m, off, 32));
    __shared__ float red[8];
    if ((threadIdx.x & 31) == 0) red[threadIdx.x >> 5] = m;
    __syncthreads();
    if (threadIdx.x == 0) {
        float mm = red[0];
        #pragma unroll
        for (int i = 1; i < 8; i++) mm = fmaxf(mm, red[i]);
        atomicMax(out_bits, __float_as_uint(mm));
    }
}

__global__ void quant_tensor_kernel(const float* __restrict__ x,
                                    const unsigned* __restrict__ amax_bits,
                                    int8_t* __restrict__ q,
                                    float* __restrict__ scale_out, size_t n4) {
    const float amax  = __uint_as_float(*amax_bits);
    const float scale = fmaxf(amax * (1.0f / QMAXF), EPSQ);
    const float inv   = 1.0f / scale;
    if (blockIdx.x == 0 && threadIdx.x == 0) *scale_out = scale;
    const float4* x4 = (const float4*)x;
    char4* q4 = (char4*)q;
    size_t stride = (size_t)gridDim.x * blockDim.x;
    for (size_t i = (size_t)blockIdx.x * blockDim.x + threadIdx.x; i < n4; i += stride) {
        float4 v = x4[i];
        char4 o;
        o.x = (signed char)(int)fminf(fmaxf(rintf(v.x * inv), -QMAXF), QMAXF);
        o.y = (signed char)(int)fminf(fmaxf(rintf(v.y * inv), -QMAXF), QMAXF);
        o.z = (signed char)(int)fminf(fmaxf(rintf(v.z * inv), -QMAXF), QMAXF);
        o.w = (signed char)(int)fminf(fmaxf(rintf(v.w * inv), -QMAXF), QMAXF);
        q4[i] = o;
    }
}

__global__ void quant_channel_kernel(const float* __restrict__ w,
                                     int8_t* __restrict__ q,
                                     float* __restrict__ sw, int ncols) {
    const int row = blockIdx.x;
    const float4* wr = (const float4*)(w + (size_t)row * ncols);
    char4* qr = (char4*)(q + (size_t)row * ncols);
    const int n4 = ncols >> 2;
    float m = 0.0f;
    for (int i = threadIdx.x; i < n4; i += blockDim.x) {
        float4 v = wr[i];
        m = fmaxf(m, fmaxf(fmaxf(fabsf(v.x), fabsf(v.y)), fmaxf(fabsf(v.z), fabsf(v.w))));
    }
    for (int off = 16; off > 0; off >>= 1)
        m = fmaxf(m, __shfl_xor(m, off, 32));
    __shared__ float red[8];
    __shared__ float s_inv;
    if ((threadIdx.x & 31) == 0) red[threadIdx.x >> 5] = m;
    __syncthreads();
    if (threadIdx.x == 0) {
        float mm = red[0];
        #pragma unroll
        for (int i = 1; i < 8; i++) mm = fmaxf(mm, red[i]);
        float scale = fmaxf(mm * (1.0f / QMAXF), EPSQ);
        sw[row] = scale;
        s_inv = 1.0f / scale;
    }
    __syncthreads();
    const float inv = s_inv;
    for (int i = threadIdx.x; i < n4; i += blockDim.x) {
        float4 v = wr[i];
        char4 o;
        o.x = (signed char)(int)fminf(fmaxf(rintf(v.x * inv), -QMAXF), QMAXF);
        o.y = (signed char)(int)fminf(fmaxf(rintf(v.y * inv), -QMAXF), QMAXF);
        o.z = (signed char)(int)fminf(fmaxf(rintf(v.z * inv), -QMAXF), QMAXF);
        o.w = (signed char)(int)fminf(fmaxf(rintf(v.w * inv), -QMAXF), QMAXF);
        qr[i] = o;
    }
}

#define TILE_M 128
#define TILE_N 128
#define TILE_K 64
#define LDSTR  80

template<bool FUSE_GELU>
__global__ __launch_bounds__(256)
void gemm_i8_wmma(const int8_t* __restrict__ A,
                  const int8_t* __restrict__ B,
                  const float* __restrict__ sx_ptr,
                  const float* __restrict__ sw,
                  const float* __restrict__ bias,
                  float* __restrict__ out,
                  unsigned* __restrict__ amax_bits,
                  int M, int N, int K) {
    __shared__ alignas(16) char lA[2][TILE_M * LDSTR];
    __shared__ alignas(16) char lB[2][TILE_N * LDSTR];
    __shared__ float sred[8];

    const int tid  = threadIdx.x;
    const int lane = tid & 31;
    const int wave = tid >> 5;
    const int wm   = wave & 3;
    const int wn   = wave >> 2;
    const int m0   = blockIdx.x * TILE_M;
    const int n0   = blockIdx.y * TILE_N;

    const int r0 = tid >> 2;
    const int r1 = r0 + 64;
    const int c0 = (tid & 3) << 4;
    const size_t aOff0 = (size_t)(m0 + r0) * K + c0;
    const size_t aOff1 = (size_t)(m0 + r1) * K + c0;
    const size_t bOff0 = (size_t)(n0 + r0) * K + c0;
    const size_t bOff1 = (size_t)(n0 + r1) * K + c0;

    unsigned ldsA0[2], ldsA1[2], ldsB0[2], ldsB1[2];
    #pragma unroll
    for (int p = 0; p < 2; p++) {
        ldsA0[p] = (unsigned)(uintptr_t)(&lA[p][r0 * LDSTR + c0]);
        ldsA1[p] = (unsigned)(uintptr_t)(&lA[p][r1 * LDSTR + c0]);
        ldsB0[p] = (unsigned)(uintptr_t)(&lB[p][r0 * LDSTR + c0]);
        ldsB1[p] = (unsigned)(uintptr_t)(&lB[p][r1 * LDSTR + c0]);
    }

    auto issue_tile = [&](int p, int k0) {
        const int8_t* ga0 = A + aOff0 + k0;
        const int8_t* ga1 = A + aOff1 + k0;
        const int8_t* gb0 = B + bOff0 + k0;
        const int8_t* gb1 = B + bOff1 + k0;
        asm volatile("global_load_async_to_lds_b128 %0, %1, off"
                     :: "v"(ldsA0[p]), "v"(ga0) : "memory");
        asm volatile("global_load_async_to_lds_b128 %0, %1, off"
                     :: "v"(ldsA1[p]), "v"(ga1) : "memory");
        asm volatile("global_load_async_to_lds_b128 %0, %1, off"
                     :: "v"(ldsB0[p]), "v"(gb0) : "memory");
        asm volatile("global_load_async_to_lds_b128 %0, %1, off"
                     :: "v"(ldsB1[p]), "v"(gb1) : "memory");
    };

    v8i_t acc[2][4];
    const v8i_t vzero = {0, 0, 0, 0, 0, 0, 0, 0};
    #pragma unroll
    for (int i = 0; i < 2; i++)
        #pragma unroll
        for (int j = 0; j < 4; j++) acc[i][j] = vzero;

    const int h   = lane >> 4;
    const int l15 = lane & 15;
    const int aFragOff0 = (wm * 32 + 0 + l15) * LDSTR + h * 8;
    const int aFragOff1 = (wm * 32 + 16 + l15) * LDSTR + h * 8;
    const int bFragOff  = (wn * 64 + l15) * LDSTR + h * 8;

    issue_tile(0, 0);
    int p = 0;
    for (int k0 = 0; k0 < K; k0 += TILE_K) {
        const bool has_next = (k0 + TILE_K) < K;
        if (has_next) {
            issue_tile(p ^ 1, k0 + TILE_K);
            if (k0 + 2 * TILE_K < K) {
                __builtin_prefetch(A + aOff0 + k0 + 2 * TILE_K, 0, 0);
                __builtin_prefetch(B + bOff0 + k0 + 2 * TILE_K, 0, 0);
            }
            asm volatile("s_wait_asynccnt 0x4" ::: "memory");
        } else {
            asm volatile("s_wait_asynccnt 0x0" ::: "memory");
        }
        __syncthreads();

        const char* aT = lA[p];
        const char* bT = lB[p];

        v8i_t af[2];
        {
            int2 p0 = *(const int2*)(aT + aFragOff0 + 0);
            int2 p1 = *(const int2*)(aT + aFragOff0 + 16);
            int2 p2 = *(const int2*)(aT + aFragOff0 + 32);
            int2 p3 = *(const int2*)(aT + aFragOff0 + 48);
            v8i_t t = {p0.x, p0.y, p1.x, p1.y, p2.x, p2.y, p3.x, p3.y};
            af[0] = t;
        }
        {
            int2 p0 = *(const int2*)(aT + aFragOff1 + 0);
            int2 p1 = *(const int2*)(aT + aFragOff1 + 16);
            int2 p2 = *(const int2*)(aT + aFragOff1 + 32);
            int2 p3 = *(const int2*)(aT + aFragOff1 + 48);
            v8i_t t = {p0.x, p0.y, p1.x, p1.y, p2.x, p2.y, p3.x, p3.y};
            af[1] = t;
        }
        v8i_t bf[4];
        #pragma unroll
        for (int j = 0; j < 4; j++) {
            const char* bb = bT + bFragOff + j * 16 * LDSTR;
            int2 q0 = *(const int2*)(bb + 0);
            int2 q1 = *(const int2*)(bb + 16);
            int2 q2 = *(const int2*)(bb + 32);
            int2 q3 = *(const int2*)(bb + 48);
            v8i_t t = {q0.x, q0.y, q1.x, q1.y, q2.x, q2.y, q3.x, q3.y};
            bf[j] = t;
        }

        #pragma unroll
        for (int i = 0; i < 2; i++)
            #pragma unroll
            for (int j = 0; j < 4; j++)
                acc[i][j] = __builtin_amdgcn_wmma_i32_16x16x64_iu8(
                     true, af[i],  true, bf[j],
                    acc[i][j],  false,  false);

        __syncthreads();
        p ^= 1;
    }

    const float sxv = *sx_ptr;
    float lmax = 0.0f;
    #pragma unroll
    for (int i = 0; i < 2; i++) {
        const int rowb = m0 + wm * 32 + i * 16 + h * 8;
        #pragma unroll
        for (int j = 0; j < 4; j++) {
            const int col = n0 + wn * 64 + j * 16 + l15;
            const float cs = sxv * sw[col];
            const float bv = bias[col];
            #pragma unroll
            for (int v = 0; v < 8; v++) {
                float val = (float)acc[i][j][v] * cs + bv;
                if (FUSE_GELU) {
                    val = gelu_exact(val);
                    lmax = fmaxf(lmax, fabsf(val));
                }
                out[(size_t)(rowb + v) * N + col] = val;
            }
        }
    }
    if (FUSE_GELU) {
        for (int off = 16; off > 0; off >>= 1)
            lmax = fmaxf(lmax, __shfl_xor(lmax, off, 32));
        if (lane == 0) sred[wave] = lmax;
        __syncthreads();
        if (tid == 0) {
            float mm = sred[0];
            #pragma unroll
            for (int i = 1; i < 8; i++) mm = fmaxf(mm, sred[i]);
            atomicMax(amax_bits, __float_as_uint(mm));
        }
    }
}

extern "C" void kernel_launch(void* const* d_in, const int* in_sizes, int n_in,
                              void* d_out, int out_size, void* d_ws, size_t ws_size,
                              hipStream_t stream) {
    (void)in_sizes; (void)n_in; (void)out_size; (void)ws_size;
    const float* x  = (const float*)d_in[0];
    const float* w1 = (const float*)d_in[1];
    const float* b1 = (const float*)d_in[2];
    const float* w2 = (const float*)d_in[3];
    const float* b2 = (const float*)d_in[4];
    float* out = (float*)d_out;

    const int Mtok = 4 * 2048;
    const int H = 2048;
    const int I = 8192;

    char* ws = (char*)d_ws;
    size_t off = 0;
    auto carve = [&](size_t bytes) -> char* {
        char* p = ws + off;
        off = (off + bytes + 255) & ~(size_t)255;
        return p;
    };
    int8_t*   qx   = (int8_t*)carve((size_t)Mtok * H);
    int8_t*   qw1  = (int8_t*)carve((size_t)I * H);
    int8_t*   qh   = (int8_t*)carve((size_t)Mtok * I);
    int8_t*   qw2  = (int8_t*)carve((size_t)H * I);
    float*    hbuf = (float*)carve((size_t)Mtok * I * sizeof(float));
    float*    sw1  = (float*)carve((size_t)I * sizeof(float));
    float*    sw2  = (float*)carve((size_t)H * sizeof(float));
    float*    scales = (float*)carve(64);
    unsigned* amax   = (unsigned*)carve(64);

    hipMemsetAsync(amax, 0, 2 * sizeof(unsigned), stream);

    absmax_kernel<<<512, 256, 0, stream>>>(x, ((size_t)Mtok * H) >> 2, amax + 0);
    quant_tensor_kernel<<<1024, 256, 0, stream>>>(x, amax + 0, qx, scales + 0,
                                                  ((size_t)Mtok * H) >> 2);
    quant_channel_kernel<<<I, 256, 0, stream>>>(w1, qw1, sw1, H);
    quant_channel_kernel<<<H, 256, 0, stream>>>(w2, qw2, sw2, I);

    dim3 g1(Mtok / TILE_M, I / TILE_N);
    gemm_i8_wmma<true><<<g1, 256, 0, stream>>>(qx, qw1, scales + 0, sw1, b1,
                                               hbuf, amax + 1, Mtok, I, H);

    quant_tensor_kernel<<<2048, 256, 0, stream>>>(hbuf, amax + 1, qh, scales + 1,
                                                  ((size_t)Mtok * I) >> 2);

    dim3 g2(Mtok / TILE_M, H / TILE_N);
    gemm_i8_wmma<false><<<g2, 256, 0, stream>>>(qh, qw2, scales + 1, sw2, b2,
                                                out, nullptr, Mtok, H, I);
}
